// MultiInterestExtractor_48473000903425
// MI455X (gfx1250) — matrix operand order repaired and checked by reading, not compiled, not run on hardware
//
#include <hip/hip_runtime.h>
#include <hip/hip_bf16.h>

typedef __attribute__((ext_vector_type(16))) __bf16 v16bf;
typedef __attribute__((ext_vector_type(8)))  float  v8f;

#define L_DIM   200
#define D_DIM   256
#define K_CAP   8
#define LDSTR   264          // padded LDS row stride in bf16 elems (breaks bank conflicts)
#define NEG_INF (-1e9f)
#define NOUT    7            // ceil(200/32) outer M-tiles of 32 rows

// LDS layout (bytes):
//   W  bf16 [256][264] = 135168
//   U  bf16 [200][264] = 105600
//   X  bf16 [ 32][264] =  16896
//   blog f32 [8][200]  =   6400
//   warr f32 [8][200]  =   6400
//   zarr f32 [8][256]  =   8192
#define SMEM_BYTES (135168 + 105600 + 16896 + 6400 + 6400 + 8192)

__device__ __forceinline__ float wave_max(float v) {
#pragma unroll
  for (int o = 16; o > 0; o >>= 1) v = fmaxf(v, __shfl_xor(v, o, 32));
  return v;
}
__device__ __forceinline__ float wave_sum(float v) {
#pragma unroll
  for (int o = 16; o > 0; o >>= 1) v += __shfl_xor(v, o, 32);
  return v;
}

// convert float4 -> 4 packed bf16 and store 8 bytes to LDS
__device__ __forceinline__ void pack4_store(__bf16* dst, float4 v) {
  union { __bf16 h[4]; uint2 u; } p;
  p.h[0] = (__bf16)v.x; p.h[1] = (__bf16)v.y;
  p.h[2] = (__bf16)v.z; p.h[3] = (__bf16)v.w;
  *(uint2*)dst = p.u;
}

__global__ __launch_bounds__(256)
void mie_routing_kernel(const float* __restrict__ x,      // [B,L,D]
                        const float* __restrict__ W,      // [D,D]
                        const float* __restrict__ binit,  // [B,K,L]
                        const int*   __restrict__ mask,   // [B,L]
                        float* __restrict__ out)          // [B,K,D]
{
  extern __shared__ char smem[];
  __bf16* Wl   = (__bf16*)smem;                 // [256][LDSTR]
  __bf16* Ul   = Wl + 256 * LDSTR;              // [200][LDSTR]  u_hat
  __bf16* Xl   = Ul + L_DIM * LDSTR;            // [32][LDSTR]   x tile (2 M-tiles)
  float*  blog = (float*)(Xl + 32 * LDSTR);     // [8][200]
  float*  warr = blog + K_CAP * L_DIM;          // [8][200]
  float*  zarr = warr + K_CAP * L_DIM;          // [8][256]  (z, then c in place)

  const int  t    = threadIdx.x;
  const int  lane = t & 31;
  const int  w    = t >> 5;                     // wave id 0..7
  const long b    = blockIdx.x;

  const int rbase = t >> 6;                     // this thread's row offset (0..3)
  const int dcol  = (t & 63) * 4;               // this thread's 4-wide column

  // ---- Stage W (f32 -> bf16) into LDS, vectorized float4 / packed bf16x4 ----
#pragma unroll 4
  for (int i = 0; i < 64; ++i) {
    int e = 4 * i + rbase;
    float4 v = *(const float4*)(W + (long)e * D_DIM + dcol);
    pack4_store(Wl + e * LDSTR + dcol, v);
  }

  // ---- Init masked routing logits ----
  for (int idx = t; idx < K_CAP * L_DIM; idx += 256) {
    int k = idx / L_DIM, l = idx - k * L_DIM;
    float bv = binit[(b * K_CAP + k) * L_DIM + l];
    blog[idx] = (mask[b * L_DIM + l] == 0) ? NEG_INF : bv;
  }

  // ---- Preload x tile 0 into registers (32 rows x 256 cols, f32) ----
  float4 xs[8];
  {
    const int lob = 0;
#pragma unroll
    for (int i = 0; i < 8; ++i) {
      int l = lob + 4 * i + rbase;
      xs[i] = (l < L_DIM) ? *(const float4*)(x + ((long)b * L_DIM + l) * D_DIM + dcol)
                          : make_float4(0.f, 0.f, 0.f, 0.f);
    }
  }
  __syncthreads();

  // ---- GEMM1: u_hat[l,e] = sum_d x[l,d] * W[e,d]; bf16 WMMA, f32 accum ----
  // Wave owns 2 N-tiles (e0,e1); each phase covers 2 M-tiles so B frags are reused.
  const int kg   = lane >> 4;        // K half-group (0/1) per ISA A/B layouts
  const int mrow = lane & 15;        // A-matrix row within a 16-row tile
  const int e0   = (w * 2) * 16 + (lane & 15);
  const int e1   = e0 + 16;

  for (int ot = 0; ot < NOUT; ++ot) {
    const int lob = ot * 32;

    // commit the register-staged tile (f32 -> bf16) into LDS
#pragma unroll
    for (int i = 0; i < 8; ++i)
      pack4_store(Xl + (4 * i + rbase) * LDSTR + dcol, xs[i]);
    __syncthreads();

    // immediately issue next tile's global loads (in flight during WMMA work)
    if (ot + 1 < NOUT) {
      const int lobn = lob + 32;
#pragma unroll
      for (int i = 0; i < 8; ++i) {
        int l = lobn + 4 * i + rbase;
        xs[i] = (l < L_DIM) ? *(const float4*)(x + ((long)b * L_DIM + l) * D_DIM + dcol)
                            : make_float4(0.f, 0.f, 0.f, 0.f);
      }
    }
    // deep prefetch of tile ot+2 (gfx1250 global_prefetch_b8)
    if (ot + 2 < NOUT) {
      int l = lob + 64 + rbase * 8;
      if (l < L_DIM) __builtin_prefetch(x + ((long)b * L_DIM + l) * D_DIM + dcol, 0, 1);
    }

    v8f a00 = {}, a01 = {}, a10 = {}, a11 = {};
#pragma unroll
    for (int kk = 0; kk < 8; ++kk) {
      const int kbase = kk * 32;
      union { uint4 q[2]; v16bf v; } bf0, bf1, af0, af1;
      // B 32x16 bf16: lane half kg holds rows K=16*kg..16*kg+15 of column e (contig in W row e)
      bf0.q[0] = *(const uint4*)(Wl + (long)e0 * LDSTR + kbase + 16 * kg);
      bf0.q[1] = *(const uint4*)(Wl + (long)e0 * LDSTR + kbase + 16 * kg + 8);
      bf1.q[0] = *(const uint4*)(Wl + (long)e1 * LDSTR + kbase + 16 * kg);
      bf1.q[1] = *(const uint4*)(Wl + (long)e1 * LDSTR + kbase + 16 * kg + 8);
      // A 16x32 bf16: lane half kg holds K = 16*kg+0..15 (contiguous)
      af0.q[0] = *(const uint4*)(Xl + mrow * LDSTR + kbase + 16 * kg);
      af0.q[1] = *(const uint4*)(Xl + mrow * LDSTR + kbase + 16 * kg + 8);
      af1.q[0] = *(const uint4*)(Xl + (16 + mrow) * LDSTR + kbase + 16 * kg);
      af1.q[1] = *(const uint4*)(Xl + (16 + mrow) * LDSTR + kbase + 16 * kg + 8);

      a00 = __builtin_amdgcn_wmma_f32_16x16x32_bf16(false, af0.v, false, bf0.v,
                                                    (short)0, a00, false, false);
      a01 = __builtin_amdgcn_wmma_f32_16x16x32_bf16(false, af0.v, false, bf1.v,
                                                    (short)0, a01, false, false);
      a10 = __builtin_amdgcn_wmma_f32_16x16x32_bf16(false, af1.v, false, bf0.v,
                                                    (short)0, a10, false, false);
      a11 = __builtin_amdgcn_wmma_f32_16x16x32_bf16(false, af1.v, false, bf1.v,
                                                    (short)0, a11, false, false);
    }

    // C/D f32 16x16 layout: VGPR r, lanes 0-15 -> M=r, lanes 16-31 -> M=r+8
#pragma unroll
    for (int r = 0; r < 8; ++r) {
      int l0 = lob + r + 8 * kg;
      int l1 = l0 + 16;
      if (l0 < L_DIM) {
        Ul[l0 * LDSTR + e0] = (__bf16)a00[r];
        Ul[l0 * LDSTR + e1] = (__bf16)a01[r];
      }
      if (l1 < L_DIM) {
        Ul[l1 * LDSTR + e0] = (__bf16)a10[r];
        Ul[l1 * LDSTR + e1] = (__bf16)a11[r];
      }
    }
    __syncthreads();
  }

  // ---- Dynamic routing, 3 iterations, all in LDS ----
  for (int it = 0; it < 3; ++it) {
    // softmax over L for capsule k = wave id
    {
      const int k = w;
      float m = -3.4e38f;
      for (int l = lane; l < L_DIM; l += 32) m = fmaxf(m, blog[k * L_DIM + l]);
      m = wave_max(m);
      float s = 0.0f;
      for (int l = lane; l < L_DIM; l += 32) s += __expf(blog[k * L_DIM + l] - m);
      s = wave_sum(s);
      float inv = 1.0f / s;
      for (int l = lane; l < L_DIM; l += 32)
        warr[k * L_DIM + l] = __expf(blog[k * L_DIM + l] - m) * inv;
    }
    __syncthreads();

    // z[k,dd] = sum_l w[k,l] * u_hat[l,dd]; thread t owns dd = t for all k
    {
      float acc[K_CAP];
#pragma unroll
      for (int k = 0; k < K_CAP; ++k) acc[k] = 0.0f;
      const int dd = t;
      for (int l = 0; l < L_DIM; ++l) {
        float u = (float)Ul[l * LDSTR + dd];
#pragma unroll
        for (int k = 0; k < K_CAP; ++k) acc[k] += warr[k * L_DIM + l] * u;
      }
#pragma unroll
      for (int k = 0; k < K_CAP; ++k) zarr[k * D_DIM + dd] = acc[k];
    }
    __syncthreads();

    // squash in place: c = nsq/(1+nsq) * z/sqrt(nsq+1e-8); wave per k
    {
      const int k = w;
      float nsq = 0.0f;
#pragma unroll
      for (int j = 0; j < 8; ++j) {
        float zv = zarr[k * D_DIM + lane + 32 * j];
        nsq += zv * zv;
      }
      nsq = wave_sum(nsq);
      float scale = nsq / ((1.0f + nsq) * sqrtf(nsq + 1e-8f));
#pragma unroll
      for (int j = 0; j < 8; ++j) zarr[k * D_DIM + lane + 32 * j] *= scale;
    }
    __syncthreads();

    // b[k,l] += sum_dd c[k,dd] * u_hat[l,dd]; thread t owns row l = t (t < 200)
    if (it < 2) {
      if (t < L_DIM) {
        const int l = t;
        float acc[K_CAP];
#pragma unroll
        for (int k = 0; k < K_CAP; ++k) acc[k] = 0.0f;
        for (int dd = 0; dd < D_DIM; ++dd) {
          float u = (float)Ul[l * LDSTR + dd];
#pragma unroll
          for (int k = 0; k < K_CAP; ++k) acc[k] += zarr[k * D_DIM + dd] * u;
        }
#pragma unroll
        for (int k = 0; k < K_CAP; ++k) blog[k * L_DIM + l] += acc[k];
      }
      __syncthreads();
    }
  }

  // ---- Write c (f32) ----
  for (int idx = t; idx < K_CAP * D_DIM; idx += 256)
    out[(long)b * (K_CAP * D_DIM) + idx] = zarr[idx];
}

extern "C" void kernel_launch(void* const* d_in, const int* in_sizes, int n_in,
                              void* d_out, int out_size, void* d_ws, size_t ws_size,
                              hipStream_t stream) {
  const float* x     = (const float*)d_in[0];   // behavior_embeds [B,L,D]
  const float* W     = (const float*)d_in[1];   // [D,D]
  const float* binit = (const float*)d_in[2];   // [B,K,L]
  const int*   mask  = (const int*)d_in[3];     // [B,L]
  float* out = (float*)d_out;

  const int B = in_sizes[0] / (L_DIM * D_DIM);  // 2048

  (void)hipFuncSetAttribute((const void*)mie_routing_kernel,
                            hipFuncAttributeMaxDynamicSharedMemorySize, SMEM_BYTES);

  mie_routing_kernel<<<dim3(B), dim3(256), SMEM_BYTES, stream>>>(x, W, binit, mask, out);
}
